// QLinear_35777077576579
// MI455X (gfx1250) — 8-bit matrix operand order repaired and checked by reading, compiled, not run on hardware
//
#include <hip/hip_runtime.h>

typedef __attribute__((ext_vector_type(8))) int v8i;

#define QMAXF 127.0f

#if __has_builtin(__builtin_amdgcn_global_load_async_to_lds_b128)
#define HAVE_ASYNC_LDS 1
typedef int v4i_ __attribute__((vector_size(4 * sizeof(int))));
typedef __attribute__((address_space(1))) v4i_* gp16_t;
typedef __attribute__((address_space(3))) v4i_* lp16_t;
__device__ __forceinline__ void async_cp16(const signed char* g, signed char* l) {
  __builtin_amdgcn_global_load_async_to_lds_b128((gp16_t)g, (lp16_t)l, 0, 0);
}
#if __has_builtin(__builtin_amdgcn_s_wait_asynccnt)
#define WAIT_ASYNC() __builtin_amdgcn_s_wait_asynccnt(0)
#else
#define WAIT_ASYNC() asm volatile("s_wait_asynccnt 0x0" ::: "memory")
#endif
#else
#define HAVE_ASYNC_LDS 0
#endif

__global__ __launch_bounds__(256) void quantize_rows(
    const float* __restrict__ x, signed char* __restrict__ q,
    float* __restrict__ iscale, int K) {
  const int row = blockIdx.x;
  const float* xr = x + (size_t)row * K;

  float amax = 0.0f;
  for (int i = threadIdx.x * 4; i < K; i += 256 * 4) {
    float4 v = *(const float4*)(xr + i);
    amax = fmaxf(amax, fmaxf(fmaxf(fabsf(v.x), fabsf(v.y)),
                             fmaxf(fabsf(v.z), fabsf(v.w))));
  }
  #pragma unroll
  for (int off = 16; off > 0; off >>= 1)
    amax = fmaxf(amax, __shfl_xor(amax, off, 32));

  __shared__ float red[8];
  __shared__ float s_inv;
  if ((threadIdx.x & 31) == 0) red[threadIdx.x >> 5] = amax;
  __syncthreads();
  if (threadIdx.x == 0) {
    float m = red[0];
    #pragma unroll
    for (int i = 1; i < 8; ++i) m = fmaxf(m, red[i]);
    float isc = m / QMAXF;
    iscale[row] = isc;
    s_inv = 1.0f / fmaxf(isc, 1e-12f);
  }
  __syncthreads();
  const float inv = s_inv;

  signed char* qr = q + (size_t)row * K;
  for (int i = threadIdx.x * 4; i < K; i += 256 * 4) {
    float4 v = *(const float4*)(xr + i);
    int a = (int)rintf(v.x * inv); a = min(max(a, -127), 127);
    int b = (int)rintf(v.y * inv); b = min(max(b, -127), 127);
    int c = (int)rintf(v.z * inv); c = min(max(c, -127), 127);
    int d = (int)rintf(v.w * inv); d = min(max(d, -127), 127);
    int packed = (a & 0xff) | ((b & 0xff) << 8) | ((c & 0xff) << 16) | (d << 24);
    *(int*)(qr + i) = packed;
  }
}

__global__ __launch_bounds__(256) void transpose_w(
    const signed char* __restrict__ w, signed char* __restrict__ wt,
    int K, int N) {
  __shared__ signed char tile[32][36];
  const int n0 = blockIdx.x * 32;
  const int k0 = blockIdx.y * 32;
  const int t = threadIdx.x;
  const int r = t >> 3;
  const int c = t & 7;

  int v = *(const int*)(w + (size_t)(k0 + r) * N + n0 + 4 * c);
  tile[r][4 * c + 0] = (signed char)(v);
  tile[r][4 * c + 1] = (signed char)(v >> 8);
  tile[r][4 * c + 2] = (signed char)(v >> 16);
  tile[r][4 * c + 3] = (signed char)(v >> 24);
  __syncthreads();

  int o = (tile[4 * c + 0][r] & 0xff)
        | ((tile[4 * c + 1][r] & 0xff) << 8)
        | ((tile[4 * c + 2][r] & 0xff) << 16)
        | ((tile[4 * c + 3][r] & 0xff) << 24);
  *(int*)(wt + (size_t)(n0 + r) * K + k0 + 4 * c) = o;
}

#define LDSP 80

__global__ __launch_bounds__(256) void gemm_iu8(
    const signed char* __restrict__ q,
    const signed char* __restrict__ wt,
    const float* __restrict__ iscale,
    const float* __restrict__ wscale,
    const float* __restrict__ bias,
    float* __restrict__ out,
    int M, int N, int K) {
  __shared__ signed char As[2][128 * LDSP];
  __shared__ signed char Bs[2][256 * LDSP];

  const int lane = threadIdx.x & 31;
  const int wid  = threadIdx.x >> 5;

  const int bm = blockIdx.y * 128;
  const int bn = blockIdx.x * 256;
  const int wm = 64 * (wid >> 2);
  const int wn = 64 * (wid & 3);

  const int lr = threadIdx.x >> 2;
  const int lc = (threadIdx.x & 3) * 16;

  const signed char* qa0 = q  + (size_t)(bm + lr) * K + lc;
  const signed char* qa1 = q  + (size_t)(bm + lr + 64) * K + lc;
  const signed char* wb0 = wt + (size_t)(bn + lr) * K + lc;
  const signed char* wb1 = wt + (size_t)(bn + lr + 64) * K + lc;
  const signed char* wb2 = wt + (size_t)(bn + lr + 128) * K + lc;
  const signed char* wb3 = wt + (size_t)(bn + lr + 192) * K + lc;

  v8i acc[4][4] = {};

  const int arow  = lane & 15;
  const int khalf = (lane >> 4) * 16;
  const int kbse  = (lane >> 4) * 16;
  const int bcol  = lane & 15;

  const int nsteps = K / 64;

#if HAVE_ASYNC_LDS
  async_cp16(qa0, &As[0][lr * LDSP + lc]);
  async_cp16(qa1, &As[0][(lr + 64) * LDSP + lc]);
  async_cp16(wb0, &Bs[0][lr * LDSP + lc]);
  async_cp16(wb1, &Bs[0][(lr + 64) * LDSP + lc]);
  async_cp16(wb2, &Bs[0][(lr + 128) * LDSP + lc]);
  async_cp16(wb3, &Bs[0][(lr + 192) * LDSP + lc]);
  WAIT_ASYNC();
  __syncthreads();
#else
  int4 a0 = *(const int4*)(qa0);
  int4 a1 = *(const int4*)(qa1);
  int4 b0 = *(const int4*)(wb0);
  int4 b1 = *(const int4*)(wb1);
  int4 b2 = *(const int4*)(wb2);
  int4 b3 = *(const int4*)(wb3);
  *(int4*)(&As[0][lr * LDSP + lc])          = a0;
  *(int4*)(&As[0][(lr + 64) * LDSP + lc])   = a1;
  *(int4*)(&Bs[0][lr * LDSP + lc])          = b0;
  *(int4*)(&Bs[0][(lr + 64) * LDSP + lc])   = b1;
  *(int4*)(&Bs[0][(lr + 128) * LDSP + lc])  = b2;
  *(int4*)(&Bs[0][(lr + 192) * LDSP + lc])  = b3;
  __syncthreads();
#endif

  for (int kb = 0; kb < nsteps; ++kb) {
    const int buf = kb & 1;
    const int knext = (kb + 1) * 64;

    if (kb + 1 < nsteps) {
      const int nb = buf ^ 1;
#if HAVE_ASYNC_LDS
      async_cp16(qa0 + knext, &As[nb][lr * LDSP + lc]);
      async_cp16(qa1 + knext, &As[nb][(lr + 64) * LDSP + lc]);
      async_cp16(wb0 + knext, &Bs[nb][lr * LDSP + lc]);
      async_cp16(wb1 + knext, &Bs[nb][(lr + 64) * LDSP + lc]);
      async_cp16(wb2 + knext, &Bs[nb][(lr + 128) * LDSP + lc]);
      async_cp16(wb3 + knext, &Bs[nb][(lr + 192) * LDSP + lc]);
#else
      a0 = *(const int4*)(qa0 + knext);
      a1 = *(const int4*)(qa1 + knext);
      b0 = *(const int4*)(wb0 + knext);
      b1 = *(const int4*)(wb1 + knext);
      b2 = *(const int4*)(wb2 + knext);
      b3 = *(const int4*)(wb3 + knext);
#endif
      if (kb + 2 < nsteps) {
        __builtin_prefetch(qa0 + knext + 64, 0, 1);
        __builtin_prefetch(wb0 + knext + 64, 0, 1);
        __builtin_prefetch(wb2 + knext + 64, 0, 1);
      }
    }

    v8i afrag[4];
    #pragma unroll
    for (int mt = 0; mt < 4; ++mt) {
      const signed char* ap = &As[buf][(wm + 16 * mt + arow) * LDSP + khalf];
      int2 d0 = *(const int2*)(ap);
      int2 d1 = *(const int2*)(ap + 8);
      int2 d2 = *(const int2*)(ap + 32);
      int2 d3 = *(const int2*)(ap + 40);
      v8i a;
      a[0] = d0.x; a[1] = d0.y;
      a[2] = d1.x; a[3] = d1.y;
      a[4] = d2.x; a[5] = d2.y;
      a[6] = d3.x; a[7] = d3.y;
      afrag[mt] = a;
    }

    const signed char* bp0 = &Bs[buf][(wn + bcol) * LDSP + kbse];
    int4 f0 = *(const int4*)(bp0);
    int4 f1 = *(const int4*)(bp0 + 32);
    #pragma unroll
    for (int nt = 0; nt < 4; ++nt) {
      v8i b;
      b[0] = f0.x; b[1] = f0.y; b[2] = f0.z; b[3] = f0.w;
      b[4] = f1.x; b[5] = f1.y; b[6] = f1.z; b[7] = f1.w;
      if (nt < 3) {
        const signed char* bp = &Bs[buf][(wn + 16 * (nt + 1) + bcol) * LDSP + kbse];
        f0 = *(const int4*)(bp);
        f1 = *(const int4*)(bp + 32);
      }
      #pragma unroll
      for (int mt = 0; mt < 4; ++mt) {
        acc[mt][nt] = __builtin_amdgcn_wmma_i32_16x16x64_iu8(
            true, afrag[mt], true, b, acc[mt][nt], false, false);
      }
    }

    if (kb + 1 < nsteps) {
#if HAVE_ASYNC_LDS
      WAIT_ASYNC();
      __syncthreads();
#else
      __syncthreads();
      const int nb = buf ^ 1;
      *(int4*)(&As[nb][lr * LDSP + lc])         = a0;
      *(int4*)(&As[nb][(lr + 64) * LDSP + lc])  = a1;
      *(int4*)(&Bs[nb][lr * LDSP + lc])         = b0;
      *(int4*)(&Bs[nb][(lr + 64) * LDSP + lc])  = b1;
      *(int4*)(&Bs[nb][(lr + 128) * LDSP + lc]) = b2;
      *(int4*)(&Bs[nb][(lr + 192) * LDSP + lc]) = b3;
      __syncthreads();
#endif
    }
  }

  const int nOff = lane & 15;
  const int mOff = (lane >> 4) * 8;
  #pragma unroll
  for (int mt = 0; mt < 4; ++mt) {
    #pragma unroll
    for (int nt = 0; nt < 4; ++nt) {
      const int n = bn + wn + 16 * nt + nOff;
      const float wsc = wscale[n];
      const float bia = bias[n];
      #pragma unroll
      for (int r = 0; r < 8; ++r) {
        const int m = bm + wm + 16 * mt + mOff + r;
        out[(size_t)m * N + n] = (float)acc[mt][nt][r] * iscale[m] * wsc + bia;
      }
    }
  }
}

extern "C" void kernel_launch(void* const* d_in, const int* in_sizes, int n_in,
                              void* d_out, int out_size, void* d_ws, size_t ws_size,
                              hipStream_t stream) {
  const float*       inp      = (const float*)d_in[0];
  const signed char* qweight  = (const signed char*)d_in[1];
  const float*       wscale   = (const float*)d_in[2];
  const float*       bias     = (const float*)d_in[3];
  float*             out      = (float*)d_out;

  const int N = in_sizes[2];
  const int K = in_sizes[1] / N;
  const int M = in_sizes[0] / K;

  signed char* q_act  = (signed char*)d_ws;
  signed char* wt     = q_act + (size_t)M * K;
  float*       iscale = (float*)(wt + (size_t)N * K);

  quantize_rows<<<dim3(M), dim3(256), 0, stream>>>(inp, q_act, iscale, K);

  transpose_w<<<dim3(N / 32, K / 32), dim3(256), 0, stream>>>(qweight, wt, K, N);

  gemm_iu8<<<dim3(N / 256, M / 128), dim3(256), 0, stream>>>(
      q_act, wt, iscale, wscale, bias, out, M, N, K);
}
